// ModernNeuralPhysicsEngine_37117107372490
// MI455X (gfx1250) — matrix operand order repaired and checked by reading, not compiled, not run on hardware
//
#include <hip/hip_runtime.h>
#include <hip/hip_bf16.h>

typedef __attribute__((ext_vector_type(16))) _Float16 v16h;
typedef __attribute__((ext_vector_type(8)))  _Float16 v8h;
typedef __attribute__((ext_vector_type(8)))  float    v8f;

#define SSTR 200   // LDS staging row stride in halfs (400B: 16B-aligned, conflict-free b128 reads)

__device__ __forceinline__ float gelu_f(float x) {
    return 0.5f * x * (1.0f + erff(x * 0.70710678118654752440f));
}

// ---------------- phase 0: f_part / c_part (+bias folded) ----------------
__global__ __launch_bounds__(256) void prep_kernel(
    const float* __restrict__ window, const float* __restrict__ wi,
    const float* __restrict__ bi, float* __restrict__ fpart, float* __restrict__ cpart)
{
    int idx = blockIdx.x * blockDim.x + threadIdx.x;
    if (idx >= 4 * 256 * 96) return;
    int f = idx % 96;
    int n = (idx / 96) % 256;
    int b = idx / (96 * 256);
    const float* s0 = &window[((b * 3 + 0) * 256 + n) * 8];
    const float* s1 = &window[((b * 3 + 1) * 256 + n) * 8];
    float fs = 0.0f, cs = bi[f];
#pragma unroll
    for (int k = 0; k < 8; k++) {
        fs += s0[k] * wi[k * 96 + f]        + s1[k] * wi[(8 + k) * 96 + f];
        cs += s0[k] * wi[(16 + k) * 96 + f] + s1[k] * wi[(24 + k) * 96 + f];
    }
    fpart[idx] = fs;
    cpart[idx] = cs;
}

// ---------------- phase 0b: pack encoder weights into WMMA B-fragments ----------------
// B-frag element e of lane l holds W[32*kc + (l<16 ? 0 : 8) + (e<8 ? e : 8+e)][16*t + (l&15)], lane-major 32B/lane.
__global__ __launch_bounds__(256) void pack_kernel(
    const float* __restrict__ w1_0, const float* __restrict__ w2_0,
    const float* __restrict__ w1_1, const float* __restrict__ w2_1,
    const float* __restrict__ w1_2, const float* __restrict__ w2_2,
    _Float16* __restrict__ wt)
{
    int idx = blockIdx.x * blockDim.x + threadIdx.x;
    const int total = 3 * 72 * 512;
    if (idx >= total) return;
    int blk  = idx / (72 * 512);
    int rrem = idx % (72 * 512);
    int frag = rrem / 512;
    int within = rrem % 512;
    int lane = within / 16;
    int e    = within % 16;
    const float* w1 = (blk == 0) ? w1_0 : ((blk == 1) ? w1_1 : w1_2);
    const float* w2 = (blk == 0) ? w2_0 : ((blk == 1) ? w2_1 : w2_2);
    int Koff = ((lane < 16) ? 0 : 8) + ((e < 8) ? e : (8 + e));
    float val;
    if (frag < 36) {                       // fc1: (96 x 192)
        int t = frag / 3, kc = frag % 3;
        val = w1[(32 * kc + Koff) * 192 + 16 * t + (lane & 15)];
    } else {                               // fc2: (192 x 96)
        int f2 = frag - 36;
        int t = f2 / 6, kc = f2 % 6;
        val = w2[(32 * kc + Koff) * 96 + 16 * t + (lane & 15)];
    }
    wt[idx] = (_Float16)val;
}

// ---------------- main fused encoder + masked aggregation ----------------
struct EncParams {
    const float* g[3];
    const float* lb[3];
    const float* b1[3];
    const float* b2[3];
};

// 512 blocks x 8 waves; each wave = one j-slice (4 j-tiles = 64 js) of one (b,i).
__global__ __launch_bounds__(256) void enc_kernel(
    const float* __restrict__ fpart, const float* __restrict__ cpart,
    const _Float16* __restrict__ wt, const float* __restrict__ window,
    EncParams P, float* __restrict__ aggp)
{
    __shared__ _Float16 stage[8][16 * SSTR];   // 50 KB staging
    const int lane = threadIdx.x & 31;
    const int wid  = threadIdx.x >> 5;
    const int gw   = blockIdx.x * 8 + wid;     // 0..4095
    const int row  = gw >> 2;                  // (b,i)
    const int slice = gw & 3;
    const int b = row >> 8, i = row & 255;
    const int ln15 = lane & 15;
    const bool hi = lane >= 16;
    const int ho = hi ? 8 : 0;
    _Float16* sbuf = &stage[wid][0];

    float fr[6];
#pragma unroll
    for (int t = 0; t < 6; t++) fr[t] = fpart[(b * 256 + i) * 96 + 16 * t + ln15];

    const float px = window[((b * 3 + 1) * 256 + i) * 8 + 0];
    const float py = window[((b * 3 + 1) * 256 + i) * 8 + 1];

    float acc[6];
#pragma unroll
    for (int t = 0; t < 6; t++) acc[t] = 0.0f;

#pragma unroll 1
    for (int jt = slice * 4; jt < slice * 4 + 4; ++jt) {
        const int jb = jt * 16;

        // each lane computes the 8 row-masks it will actually use (rows M = v + ho)
        float mv[8];
#pragma unroll
        for (int v = 0; v < 8; v++) {
            int j = jb + v + ho;
            float dx = window[((b * 3 + 1) * 256 + j) * 8 + 0] - px;
            float dy = window[((b * 3 + 1) * 256 + j) * 8 + 1] - py;
            mv[v] = ((dx * dx + dy * dy) <= 0.25f && (j != i)) ? 1.0f : 0.0f;
        }

        // x0 = gelu(f_i + c_j(+bias)) built in WMMA C layout
        v8f x[6];
#pragma unroll
        for (int t = 0; t < 6; t++) {
#pragma unroll
            for (int v = 0; v < 8; v++) {
                float c = cpart[(b * 256 + jb + v + ho) * 96 + 16 * t + ln15];
                x[t][v] = gelu_f(fr[t] + c);
            }
        }

#pragma unroll
        for (int blk = 0; blk < 3; ++blk) {
            // ---- LayerNorm over 96 features (per row) ----
            float mu[8], rs[8];
#pragma unroll
            for (int v = 0; v < 8; v++) {
                float s = 0.0f;
#pragma unroll
                for (int t = 0; t < 6; t++) s += x[t][v];
                s += __shfl_xor(s, 1, 32);
                s += __shfl_xor(s, 2, 32);
                s += __shfl_xor(s, 4, 32);
                s += __shfl_xor(s, 8, 32);
                mu[v] = s * (1.0f / 96.0f);
            }
#pragma unroll
            for (int v = 0; v < 8; v++) {
                float s2 = 0.0f;
#pragma unroll
                for (int t = 0; t < 6; t++) { float d = x[t][v] - mu[v]; s2 += d * d; }
                s2 += __shfl_xor(s2, 1, 32);
                s2 += __shfl_xor(s2, 2, 32);
                s2 += __shfl_xor(s2, 4, 32);
                s2 += __shfl_xor(s2, 8, 32);
                rs[v] = rsqrtf(s2 * (1.0f / 96.0f) + 1e-5f);
            }
            // stage h = LN(x)*g+b to LDS (f16)
#pragma unroll
            for (int t = 0; t < 6; t++) {
                float gv = P.g[blk][16 * t + ln15];
                float bv = P.lb[blk][16 * t + ln15];
#pragma unroll
                for (int v = 0; v < 8; v++) {
                    float h = (x[t][v] - mu[v]) * rs[v] * gv + bv;
                    sbuf[(v + ho) * SSTR + 16 * t + ln15] = (_Float16)h;
                }
            }
            asm volatile("s_wait_dscnt 0" ::: "memory");

            // A fragments of h (16x32 f16 layout): 2x ds_load_b128 per frag
            v16h a1[3];
#pragma unroll
            for (int kc = 0; kc < 3; kc++) {
                const _Float16* p = sbuf + ln15 * SSTR + 32 * kc + ho;
                v8h lo = *(const v8h*)p;
                v8h hh = *(const v8h*)(p + 16);
                a1[kc] = __builtin_shufflevector(lo, hh, 0,1,2,3,4,5,6,7,8,9,10,11,12,13,14,15);
            }

            const _Float16* w1b = wt + blk * 72 * 512;
            const _Float16* w2b = w1b + 36 * 512;

            // ---- fc1 (96->192): batch the 3 B-frag loads ahead of the WMMAs ----
#pragma unroll
            for (int t = 0; t < 12; t++) {
                v16h bf[3];
#pragma unroll
                for (int kc = 0; kc < 3; kc++)
                    bf[kc] = *(const v16h*)(w1b + (t * 3 + kc) * 512 + lane * 16);
                float b1v = P.b1[blk][16 * t + ln15];
                v8f accv;
#pragma unroll
                for (int v = 0; v < 8; v++) accv[v] = b1v;
#pragma unroll
                for (int kc = 0; kc < 3; kc++)
                    accv = __builtin_amdgcn_wmma_f32_16x16x32_f16(
                        false, a1[kc], false, bf[kc], (short)0, accv, false, false);
#pragma unroll
                for (int v = 0; v < 8; v++)
                    sbuf[(v + ho) * SSTR + 16 * t + ln15] = (_Float16)gelu_f(accv[v]);
            }
            asm volatile("s_wait_dscnt 0" ::: "memory");

            // A fragments of h1 (16x192)
            v16h a2[6];
#pragma unroll
            for (int kc = 0; kc < 6; kc++) {
                const _Float16* p = sbuf + ln15 * SSTR + 32 * kc + ho;
                v8h lo = *(const v8h*)p;
                v8h hh = *(const v8h*)(p + 16);
                a2[kc] = __builtin_shufflevector(lo, hh, 0,1,2,3,4,5,6,7,8,9,10,11,12,13,14,15);
            }

            // ---- fc2 (192->96): batch the 6 B-frag loads ahead of the WMMAs ----
#pragma unroll
            for (int t = 0; t < 6; t++) {
                v16h bf[6];
#pragma unroll
                for (int kc = 0; kc < 6; kc++)
                    bf[kc] = *(const v16h*)(w2b + (t * 6 + kc) * 512 + lane * 16);
                float b2v = P.b2[blk][16 * t + ln15];
                v8f accv = x[t];
#pragma unroll
                for (int v = 0; v < 8; v++) accv[v] += b2v;
#pragma unroll
                for (int kc = 0; kc < 6; kc++)
                    accv = __builtin_amdgcn_wmma_f32_16x16x32_f16(
                        false, a2[kc], false, bf[kc], (short)0, accv, false, false);
                x[t] = accv;
            }
        } // blocks

        // masked accumulation over this tile's 16 rows (each lane handles its half)
#pragma unroll
        for (int t = 0; t < 6; t++) {
#pragma unroll
            for (int v = 0; v < 8; v++) acc[t] += mv[v] * x[t][v];
        }
    } // j tiles of this slice

    // combine lane halves and write this slice's partial agg
#pragma unroll
    for (int t = 0; t < 6; t++) {
        float tot = acc[t] + __shfl_xor(acc[t], 16, 32);
        if (!hi) aggp[(slice * 1024 + b * 256 + i) * 96 + 16 * t + ln15] = tot;
    }
}

// ---------------- decoder (small, exact fp32; sums the 4 slice partials) ----------------
struct DecParams {
    const float* din_w; const float* din_b;
    const float* g[3];  const float* lb[3];
    const float* w1[3]; const float* b1[3];
    const float* w2[3]; const float* b2[3];
    const float* gf;    const float* bf;
    const float* ow;    const float* ob;
};

__global__ __launch_bounds__(96) void dec_kernel(
    const float* __restrict__ aggp, const float* __restrict__ window,
    DecParams P, float* __restrict__ out)
{
    __shared__ float din[112];
    __shared__ float sh[192];
    __shared__ float h1b[192];
    int r = blockIdx.x;
    int b = r >> 8, i = r & 255;
    int tid = threadIdx.x;

    {
        float a = 0.0f;
#pragma unroll
        for (int s = 0; s < 4; s++) a += aggp[(s * 1024 + r) * 96 + tid];
        din[tid] = a;
    }
    if (tid < 8)              din[96 + tid]        = window[((b * 3 + 0) * 256 + i) * 8 + tid];
    if (tid >= 8 && tid < 16) din[104 + (tid - 8)] = window[((b * 3 + 1) * 256 + i) * 8 + (tid - 8)];
    __syncthreads();

    float s = P.din_b[tid];
    for (int k = 0; k < 112; k++) s += din[k] * P.din_w[k * 96 + tid];
    float y = gelu_f(s);

#pragma unroll
    for (int blk = 0; blk < 3; ++blk) {
        __syncthreads();
        sh[tid] = y;
        __syncthreads();
        float mu = 0.0f;
        for (int k = 0; k < 96; k++) mu += sh[k];
        mu *= (1.0f / 96.0f);
        float var = 0.0f;
        for (int k = 0; k < 96; k++) { float d = sh[k] - mu; var += d * d; }
        var *= (1.0f / 96.0f);
        float rs = rsqrtf(var + 1e-5f);
        float h = (y - mu) * rs * P.g[blk][tid] + P.lb[blk][tid];
        __syncthreads();
        sh[tid] = h;
        __syncthreads();
        float a0 = P.b1[blk][tid], a1v = P.b1[blk][tid + 96];
        for (int k = 0; k < 96; k++) {
            float hv = sh[k];
            a0  += hv * P.w1[blk][k * 192 + tid];
            a1v += hv * P.w1[blk][k * 192 + tid + 96];
        }
        h1b[tid]      = gelu_f(a0);
        h1b[tid + 96] = gelu_f(a1v);
        __syncthreads();
        float accv = P.b2[blk][tid];
        for (int k = 0; k < 192; k++) accv += h1b[k] * P.w2[blk][k * 96 + tid];
        y = y + accv;
    }

    __syncthreads();
    sh[tid] = y;
    __syncthreads();
    float mu = 0.0f;
    for (int k = 0; k < 96; k++) mu += sh[k];
    mu *= (1.0f / 96.0f);
    float var = 0.0f;
    for (int k = 0; k < 96; k++) { float d = sh[k] - mu; var += d * d; }
    var *= (1.0f / 96.0f);
    float rs = rsqrtf(var + 1e-5f);
    float yn = (y - mu) * rs * P.gf[tid] + P.bf[tid];
    __syncthreads();
    sh[tid] = yn;
    __syncthreads();
    if (tid < 2) {
        float o = P.ob[tid];
        for (int k = 0; k < 96; k++) o += sh[k] * P.ow[k * 2 + tid];
        out[r * 2 + tid] = o;                                                    // pred_v
        out[2048 + r * 2 + tid] = window[((b * 3 + 2) * 256 + i) * 8 + 2 + tid]; // target_v
    }
}

extern "C" void kernel_launch(void* const* d_in, const int* in_sizes, int n_in,
                              void* d_out, int out_size, void* d_ws, size_t ws_size,
                              hipStream_t stream) {
    (void)in_sizes; (void)n_in; (void)out_size; (void)ws_size;

    const float* window = (const float*)d_in[0];
    const float* enc_w  = (const float*)d_in[1];
    const float* enc_bi = (const float*)d_in[2];

    EncParams EP;
    const float* enc_w1[3]; const float* enc_w2[3];
    for (int k = 0; k < 3; k++) {
        EP.g[k]   = (const float*)d_in[3 + 6 * k];
        EP.lb[k]  = (const float*)d_in[4 + 6 * k];
        enc_w1[k] = (const float*)d_in[5 + 6 * k];
        EP.b1[k]  = (const float*)d_in[6 + 6 * k];
        enc_w2[k] = (const float*)d_in[7 + 6 * k];
        EP.b2[k]  = (const float*)d_in[8 + 6 * k];
    }
    DecParams DP;
    DP.din_w = (const float*)d_in[21];
    DP.din_b = (const float*)d_in[22];
    for (int k = 0; k < 3; k++) {
        DP.g[k]  = (const float*)d_in[23 + 6 * k];
        DP.lb[k] = (const float*)d_in[24 + 6 * k];
        DP.w1[k] = (const float*)d_in[25 + 6 * k];
        DP.b1[k] = (const float*)d_in[26 + 6 * k];
        DP.w2[k] = (const float*)d_in[27 + 6 * k];
        DP.b2[k] = (const float*)d_in[28 + 6 * k];
    }
    DP.gf = (const float*)d_in[41];
    DP.bf = (const float*)d_in[42];
    DP.ow = (const float*)d_in[43];
    DP.ob = (const float*)d_in[44];

    // Workspace layout
    float* fpart = (float*)d_ws;              // 4*256*96
    float* cpart = fpart + 98304;             // 4*256*96
    float* aggp  = cpart + 98304;             // 4 slices * 4*256*96
    _Float16* wt = (_Float16*)(aggp + 4 * 98304);  // 3*72*512 halfs

    prep_kernel<<<(98304 + 255) / 256, 256, 0, stream>>>(window, enc_w, enc_bi, fpart, cpart);
    pack_kernel<<<(110592 + 255) / 256, 256, 0, stream>>>(
        enc_w1[0], enc_w2[0], enc_w1[1], enc_w2[1], enc_w1[2], enc_w2[2], wt);
    enc_kernel<<<512, 256, 0, stream>>>(fpart, cpart, wt, window, EP, aggp);
    dec_kernel<<<1024, 96, 0, stream>>>(aggp, window, DP, (float*)d_out);
}
